// GTrans_14920716387034
// MI455X (gfx1250) — compile-verified, matrix operand order checked by reading, not run on hardware
//
#include <hip/hip_runtime.h>
#include <math.h>

// ---------------------------------------------------------------------------
// Types for CDNA5 WMMA
// ---------------------------------------------------------------------------
typedef __attribute__((ext_vector_type(16))) __bf16 bf16x16;
typedef __attribute__((ext_vector_type(8)))  float  f32x8;

__device__ __forceinline__ unsigned short f2bf(float f) {
    // round-to-nearest-even f32 -> bf16
    unsigned x = __float_as_uint(f);
    unsigned r = x + 0x7fffu + ((x >> 16) & 1u);
    return (unsigned short)(r >> 16);
}
__device__ __forceinline__ unsigned fkey(float f) {
    // monotone order-preserving f32 -> u32 (for atomicMax-based segment max)
    unsigned b = __float_as_uint(f);
    return (b & 0x80000000u) ? ~b : (b | 0x80000000u);
}
__device__ __forceinline__ float funkey(unsigned k) {
    unsigned b = (k & 0x80000000u) ? (k & 0x7fffffffu) : ~k;
    return __uint_as_float(b);
}

// ---------------------------------------------------------------------------
// f32 -> bf16 conversion (A operand for WMMA)
// ---------------------------------------------------------------------------
__global__ void k_f32_to_bf16(const float* __restrict__ src,
                              unsigned short* __restrict__ dst, long n) {
    long i = (long)blockIdx.x * blockDim.x + threadIdx.x;
    if (i < n) dst[i] = f2bf(src[i]);
}

// ---------------------------------------------------------------------------
// Pack 4 weight matrices [K,d] into one bf16 buffer pre-swizzled into the
// WMMA B-fragment layout (32x16 tiles, per-lane contiguous 16 bf16).
// Fragment element (lane, j) holds B(k_local, n_local) with
//   n_local = lane & 15
//   k_local = j + 8*((j>>3) + (lane>>4))
// Linear index: ((kt*NT + nt)*32 + lane)*16 + j
// ---------------------------------------------------------------------------
__global__ void k_pack_w(const float* __restrict__ Wq, const float* __restrict__ Wk,
                         const float* __restrict__ Wv, const float* __restrict__ Ws,
                         unsigned short* __restrict__ wf, int K, int d) {
    const int Ncols = 4 * d;
    const int NT = Ncols >> 4;
    long gid = (long)blockIdx.x * blockDim.x + threadIdx.x;
    long total = (long)K * Ncols;
    if (gid >= total) return;
    int j    = (int)(gid & 15);
    int lane = (int)((gid >> 4) & 31);
    long tile = gid >> 9;                 // / (32*16)
    int nt = (int)(tile % NT);
    int kt = (int)(tile / NT);
    int n_local = lane & 15;
    int k_local = j + 8 * ((j >> 3) + (lane >> 4));
    int k    = kt * 32 + k_local;
    int ncat = nt * 16 + n_local;
    int which = ncat / d;
    int n     = ncat - which * d;
    const float* W = (which == 0) ? Wq : (which == 1) ? Wk : (which == 2) ? Wv : Ws;
    wf[gid] = f2bf(W[(size_t)k * d + n]);
}

__global__ void k_pack_bias(const float* __restrict__ bq, const float* __restrict__ bk,
                            const float* __restrict__ bv, const float* __restrict__ bs,
                            float* __restrict__ bias, int d) {
    int i = blockIdx.x * blockDim.x + threadIdx.x;
    if (i >= 4 * d) return;
    int which = i / d, n = i - which * d;
    const float* b = (which == 0) ? bq : (which == 1) ? bk : (which == 2) ? bv : bs;
    bias[i] = b[n];
}

// ---------------------------------------------------------------------------
// Fused GEMM: C[N, 4d] = A[N,K](bf16) x Wcat[K,4d](bf16, fragment-packed) + bias
// One wave per 16x64 output strip: 4 N-tiles share one A fragment per K-step
// (4x arithmetic intensity vs a single 16x16 tile), K-loop in steps of 32 via
// v_wmma_f32_16x16x32_bf16.
// ---------------------------------------------------------------------------
__global__ __launch_bounds__(256)
void k_gemm_wmma(const unsigned short* __restrict__ A,
                 const unsigned short* __restrict__ WF,
                 const float* __restrict__ bias,
                 float* __restrict__ C, int M, int K, int Ncols) {
    const int NT = Ncols >> 4;   // 16-wide column tiles
    const int NG = NT >> 2;      // groups of 4 column tiles
    const long MT = M >> 4;
    long wid = ((long)blockIdx.x * blockDim.x + threadIdx.x) >> 5;
    int lane = threadIdx.x & 31;
    if (wid >= MT * (long)NG) return;
    int ng  = (int)(wid % NG);
    long mt = wid / NG;
    const int nt0 = ng * 4;

    f32x8 acc0 = {}, acc1 = {}, acc2 = {}, acc3 = {};
    const int hl = lane >> 4;             // lane half
    const int row = (int)(mt * 16) + (lane & 15);
    const unsigned short* arow = A + (size_t)row * K + 8 * hl;
    const int KT = K >> 5;

    union AFrag { uint4 u[2]; bf16x16 v; };

    for (int kt = 0; kt < KT; ++kt) {
        AFrag af;
        // A layout (16-bit, 16x32): lanes 0-15 hold K 0..7 / 16..23,
        // lanes 16-31 hold K 8..15 / 24..31 for row M = lane&15.
        af.u[0] = *(const uint4*)(arow + kt * 32);        // K = base..base+7
        af.u[1] = *(const uint4*)(arow + kt * 32 + 16);   // K = base+16..+23
        // 4 consecutive B fragments; each fragment is 32 lanes * 16 bf16 = 512
        const unsigned short* bbase =
            WF + ((size_t)((size_t)kt * NT + nt0) * 32 + lane) * 16;
        bf16x16 b0 = *(const bf16x16*)(bbase);
        bf16x16 b1 = *(const bf16x16*)(bbase + 512);
        bf16x16 b2 = *(const bf16x16*)(bbase + 1024);
        bf16x16 b3 = *(const bf16x16*)(bbase + 1536);
        acc0 = __builtin_amdgcn_wmma_f32_16x16x32_bf16(false, af.v, false, b0,
                                                       (short)0, acc0, false, false);
        acc1 = __builtin_amdgcn_wmma_f32_16x16x32_bf16(false, af.v, false, b1,
                                                       (short)0, acc1, false, false);
        acc2 = __builtin_amdgcn_wmma_f32_16x16x32_bf16(false, af.v, false, b2,
                                                       (short)0, acc2, false, false);
        acc3 = __builtin_amdgcn_wmma_f32_16x16x32_bf16(false, af.v, false, b3,
                                                       (short)0, acc3, false, false);
    }

    // C layout: VGPR v -> M = v + 8*(lane>>4), N = lane&15
    const long mrow0 = mt * 16 + 8 * hl;
#pragma unroll
    for (int t = 0; t < 4; ++t) {
        const f32x8 acc = (t == 0) ? acc0 : (t == 1) ? acc1 : (t == 2) ? acc2 : acc3;
        const int ncol = (nt0 + t) * 16 + (lane & 15);
        const float bv = bias[ncol];
#pragma unroll
        for (int v = 0; v < 8; ++v) {
            C[(size_t)(mrow0 + v) * Ncols + ncol] = acc[v] + bv;
        }
    }
}

// ---------------------------------------------------------------------------
// Edge phase
// ---------------------------------------------------------------------------
__global__ void k_init_state(unsigned* __restrict__ mkey, float* __restrict__ denom,
                             double* __restrict__ stats, int n) {
    int i = blockIdx.x * blockDim.x + threadIdx.x;
    if (i < n) {
        mkey[i]  = fkey(-INFINITY);
        denom[i] = 0.f;
    }
    if (i < 2) stats[i] = 0.0;
}

// One wave per edge: score = dot(q[dst], k[src]) * scale; atomicMax per dst.
__global__ void k_edge_score(const float* __restrict__ qkvs, const int* __restrict__ ei,
                             float* __restrict__ scores, unsigned* __restrict__ mkey,
                             int E, int d, int stride, float scale) {
    long wid = ((long)blockIdx.x * blockDim.x + threadIdx.x) >> 5;
    int lane = threadIdx.x & 31;
    if (wid >= E) return;
    int src = ei[wid];
    int dst = ei[(long)E + wid];
    const float4* q4 = (const float4*)(qkvs + (size_t)dst * stride);       // q @ col 0
    const float4* k4 = (const float4*)(qkvs + (size_t)src * stride + d);   // k @ col d
    float acc = 0.f;
    const int d4 = d >> 2;
    for (int t = lane; t < d4; t += 32) {
        float4 a = q4[t], b = k4[t];
        acc += a.x * b.x + a.y * b.y + a.z * b.z + a.w * b.w;
    }
#pragma unroll
    for (int o = 16; o; o >>= 1) acc += __shfl_xor(acc, o, 32);
    if (lane == 0) {
        float s = acc * scale;
        scores[wid] = s;
        atomicMax(&mkey[dst], fkey(s));
    }
}

__global__ void k_edge_exp(const float* __restrict__ scores, const int* __restrict__ ei,
                           const unsigned* __restrict__ mkey, float* __restrict__ expw,
                           float* __restrict__ denom, int E) {
    long e = (long)blockIdx.x * blockDim.x + threadIdx.x;
    if (e >= E) return;
    int dst = ei[(long)E + e];
    float m = funkey(mkey[dst]);
    float w = __expf(scores[e] - m);
    expw[e] = w;
    atomicAdd(&denom[dst], w);
}

// One wave per edge: agg[dst] += alpha * v[src]  (agg is the S slice, pre-set
// to the skip connection x@Ws+bs by the GEMM).
__global__ void k_edge_agg(float* __restrict__ qkvs, const int* __restrict__ ei,
                           const float* __restrict__ expw, const float* __restrict__ denom,
                           int E, int d, int stride) {
    long wid = ((long)blockIdx.x * blockDim.x + threadIdx.x) >> 5;
    int lane = threadIdx.x & 31;
    if (wid >= E) return;
    int src = ei[wid];
    int dst = ei[(long)E + wid];
    float alpha = expw[wid] / denom[dst];
    const float4* v4 = (const float4*)(qkvs + (size_t)src * stride + 2 * d);
    float* agg      = qkvs + (size_t)dst * stride + 3 * d;
    const int d4 = d >> 2;
    for (int t = lane; t < d4; t += 32) {
        float4 vv = v4[t];
        atomicAdd(&agg[4 * t + 0], alpha * vv.x);
        atomicAdd(&agg[4 * t + 1], alpha * vv.y);
        atomicAdd(&agg[4 * t + 2], alpha * vv.z);
        atomicAdd(&agg[4 * t + 3], alpha * vv.w);
    }
}

// ---------------------------------------------------------------------------
// Graph LayerNorm (over ALL N*d entries) + ELU (+bf16 emit) / final f32 out
// ---------------------------------------------------------------------------
__global__ void k_stats(const float* __restrict__ base, int N, int d, int stride,
                        double* __restrict__ stats) {
    __shared__ double sh0[256];
    __shared__ double sh1[256];
    long total = (long)N * d;
    long gsz = (long)gridDim.x * blockDim.x;
    double s = 0.0, s2 = 0.0;
    for (long i = (long)blockIdx.x * blockDim.x + threadIdx.x; i < total; i += gsz) {
        long r = i / d;
        int  c = (int)(i - r * (long)d);
        float v = base[r * (long)stride + c];
        s += (double)v;
        s2 += (double)v * (double)v;
    }
    int t = threadIdx.x;
    sh0[t] = s; sh1[t] = s2;
    __syncthreads();
    for (int o = 128; o; o >>= 1) {
        if (t < o) { sh0[t] += sh0[t + o]; sh1[t] += sh1[t + o]; }
        __syncthreads();
    }
    if (t == 0) { atomicAdd(&stats[0], sh0[0]); atomicAdd(&stats[1], sh1[0]); }
}

__global__ void k_norm_elu_bf16(const float* __restrict__ base, int N, int d, int stride,
                                const double* __restrict__ stats,
                                const float* __restrict__ g, const float* __restrict__ be,
                                float eps, unsigned short* __restrict__ out) {
    long total = (long)N * d;
    long i = (long)blockIdx.x * blockDim.x + threadIdx.x;
    if (i >= total) return;
    long r = i / d;
    int  c = (int)(i - r * (long)d);
    double cnt = (double)total;
    double mean = stats[0] / cnt;
    double var = stats[1] / cnt - mean * mean;
    if (var < 0.0) var = 0.0;
    float istd = 1.0f / ((float)sqrt(var) + eps);
    float v = base[r * (long)stride + c];
    float y = (v - (float)mean) * istd * g[c] + be[c];
    y = (y > 0.f) ? y : expm1f(y);       // ELU(alpha=1)
    out[i] = f2bf(y);
}

__global__ void k_norm_f32(const float* __restrict__ base, int N, int d, int stride,
                           const double* __restrict__ stats,
                           const float* __restrict__ g, const float* __restrict__ be,
                           float eps, float* __restrict__ out) {
    long total = (long)N * d;
    long i = (long)blockIdx.x * blockDim.x + threadIdx.x;
    if (i >= total) return;
    long r = i / d;
    int  c = (int)(i - r * (long)d);
    double cnt = (double)total;
    double mean = stats[0] / cnt;
    double var = stats[1] / cnt - mean * mean;
    if (var < 0.0) var = 0.0;
    float istd = 1.0f / ((float)sqrt(var) + eps);
    float v = base[r * (long)stride + c];
    out[i] = (v - (float)mean) * istd * g[c] + be[c];
}

// ---------------------------------------------------------------------------
// Host launcher
// ---------------------------------------------------------------------------
extern "C" void kernel_launch(void* const* d_in, const int* in_sizes, int n_in,
                              void* d_out, int out_size, void* d_ws, size_t ws_size,
                              hipStream_t stream) {
    (void)n_in; (void)out_size; (void)ws_size;
    const int IN_DIM = 512, HID = 256, OUT = 128;
    const int N = in_sizes[0] / IN_DIM;
    const int E = in_sizes[21] / 2;
    const float EPS = 1e-5f;

    const float* x = (const float*)d_in[0];
    const int*  ei = (const int*)d_in[21];

    // workspace carve-out
    char* wsbase = (char*)d_ws;
    size_t off = 0;
    auto carve = [&](size_t bytes) -> char* {
        char* p = wsbase + off;
        off = (off + bytes + 255) & ~(size_t)255;
        return p;
    };
    float*          qkvs  = (float*)carve((size_t)N * 4 * HID * sizeof(float)); // reused L2
    unsigned short* abf   = (unsigned short*)carve((size_t)N * IN_DIM * sizeof(unsigned short));
    unsigned short* wf1   = (unsigned short*)carve((size_t)IN_DIM * 4 * HID * sizeof(unsigned short));
    unsigned short* wf2   = (unsigned short*)carve((size_t)HID * 4 * OUT * sizeof(unsigned short));
    float*          bias1 = (float*)carve(4 * HID * sizeof(float));
    float*          bias2 = (float*)carve(4 * OUT * sizeof(float));
    float*          scores = (float*)carve((size_t)E * sizeof(float));
    float*          expw   = (float*)carve((size_t)E * sizeof(float));
    unsigned*       mkey   = (unsigned*)carve((size_t)N * sizeof(unsigned));
    float*          denom  = (float*)carve((size_t)N * sizeof(float));
    double*         stats  = (double*)carve(2 * sizeof(double));

    auto blocks = [](long threads) { return (unsigned)((threads + 255) / 256); };

    // ---------------- Layer 1 ----------------
    {
        const int K = IN_DIM, d = HID, Nc = 4 * HID;
        long xcnt = (long)N * K;
        k_f32_to_bf16<<<blocks(xcnt), 256, 0, stream>>>(x, abf, xcnt);
        k_pack_w<<<blocks((long)K * Nc), 256, 0, stream>>>(
            (const float*)d_in[1], (const float*)d_in[3],
            (const float*)d_in[5], (const float*)d_in[7], wf1, K, d);
        k_pack_bias<<<blocks(Nc), 256, 0, stream>>>(
            (const float*)d_in[2], (const float*)d_in[4],
            (const float*)d_in[6], (const float*)d_in[8], bias1, d);
        long waves = ((long)N / 16) * (Nc / 64);   // 16x64 strips
        k_gemm_wmma<<<blocks(waves * 32), 256, 0, stream>>>(abf, wf1, bias1, qkvs, N, K, Nc);

        k_init_state<<<blocks(N), 256, 0, stream>>>(mkey, denom, stats, N);
        k_edge_score<<<blocks((long)E * 32), 256, 0, stream>>>(
            qkvs, ei, scores, mkey, E, d, Nc, 1.0f / sqrtf((float)d));
        k_edge_exp<<<blocks(E), 256, 0, stream>>>(scores, ei, mkey, expw, denom, E);
        k_edge_agg<<<blocks((long)E * 32), 256, 0, stream>>>(qkvs, ei, expw, denom, E, d, Nc);

        const float* agg = qkvs + 3 * d;  // S-slice holds conv output
        k_stats<<<1024, 256, 0, stream>>>(agg, N, d, Nc, stats);
        k_norm_elu_bf16<<<blocks((long)N * d), 256, 0, stream>>>(
            agg, N, d, Nc, stats, (const float*)d_in[9], (const float*)d_in[10], EPS, abf);
    }

    // ---------------- Layer 2 ----------------
    {
        const int K = HID, d = OUT, Nc = 4 * OUT;
        k_pack_w<<<blocks((long)K * Nc), 256, 0, stream>>>(
            (const float*)d_in[11], (const float*)d_in[13],
            (const float*)d_in[15], (const float*)d_in[17], wf2, K, d);
        k_pack_bias<<<blocks(Nc), 256, 0, stream>>>(
            (const float*)d_in[12], (const float*)d_in[14],
            (const float*)d_in[16], (const float*)d_in[18], bias2, d);
        long waves = ((long)N / 16) * (Nc / 64);   // 16x64 strips
        k_gemm_wmma<<<blocks(waves * 32), 256, 0, stream>>>(abf, wf2, bias2, qkvs, N, K, Nc);

        k_init_state<<<blocks(N), 256, 0, stream>>>(mkey, denom, stats, N);
        k_edge_score<<<blocks((long)E * 32), 256, 0, stream>>>(
            qkvs, ei, scores, mkey, E, d, Nc, 1.0f / sqrtf((float)d));
        k_edge_exp<<<blocks(E), 256, 0, stream>>>(scores, ei, mkey, expw, denom, E);
        k_edge_agg<<<blocks((long)E * 32), 256, 0, stream>>>(qkvs, ei, expw, denom, E, d, Nc);

        const float* agg = qkvs + 3 * d;
        k_stats<<<1024, 256, 0, stream>>>(agg, N, d, Nc, stats);
        k_norm_f32<<<blocks((long)N * d), 256, 0, stream>>>(
            agg, N, d, Nc, stats, (const float*)d_in[19], (const float*)d_in[20], EPS,
            (float*)d_out);
    }
}
